// NXTentLoss_52080773431808
// MI455X (gfx1250) — matrix operand order repaired and checked by reading, not compiled, not run on hardware
//
#include <hip/hip_runtime.h>
#include <hip/hip_bf16.h>
#include <stdint.h>

typedef __attribute__((ext_vector_type(16))) _Float16 v16h;
typedef __attribute__((ext_vector_type(8)))  _Float16 v8h;
typedef __attribute__((ext_vector_type(8)))  float    v8f;

#define N_PAIRS   4096
#define TWO_N     8192
#define DIM       256
#define INV_T     10.0f
#define NORM_EPS  1e-12f

// LDS row pitch: 512B of row data + 16B pad -> bank rotation by 4 per row.
#define LDS_PITCH_H 264   // halfs (528 bytes)

union AFrag { v16h v; v8h h[2]; };

#define WMMA_F16(A, B, C) \
    __builtin_amdgcn_wmma_f32_16x16x32_f16(false, (A), false, (B), (short)0, (C), false, false)

// ---------------------------------------------------------------------------
// Kernel 0: zero the per-row denominator accumulators (ws is poisoned 0xAA).
// ---------------------------------------------------------------------------
__global__ __launch_bounds__(256)
void nxtent_zero_kernel(float* __restrict__ p, int n) {
    int i = blockIdx.x * 256 + threadIdx.x;
    if (i < n) p[i] = 0.0f;
}

// ---------------------------------------------------------------------------
// Kernel 1: row-normalize Z = [z1; z2] in fp32, store as f16 (WMMA operand).
// ---------------------------------------------------------------------------
__global__ __launch_bounds__(256)
void nxtent_normalize_kernel(const float* __restrict__ z1,
                             const float* __restrict__ z2,
                             _Float16* __restrict__ Zh) {
    const int wave = threadIdx.x >> 5;
    const int lane = threadIdx.x & 31;
    const int row  = blockIdx.x * 8 + wave;              // 0..8191

    const float* src = (row < N_PAIRS)
                         ? (z1 + (size_t)row * DIM)
                         : (z2 + (size_t)(row - N_PAIRS) * DIM);

    const float4 va = ((const float4*)src)[lane * 2 + 0];
    const float4 vb = ((const float4*)src)[lane * 2 + 1];

    float ss = va.x*va.x + va.y*va.y + va.z*va.z + va.w*va.w
             + vb.x*vb.x + vb.y*vb.y + vb.z*vb.z + vb.w*vb.w;
    #pragma unroll
    for (int off = 1; off < 32; off <<= 1)
        ss += __shfl_xor(ss, off, 32);

    const float inv = 1.0f / fmaxf(sqrtf(ss), NORM_EPS);

    v8h h;
    h[0] = (_Float16)(va.x * inv); h[1] = (_Float16)(va.y * inv);
    h[2] = (_Float16)(va.z * inv); h[3] = (_Float16)(va.w * inv);
    h[4] = (_Float16)(vb.x * inv); h[5] = (_Float16)(vb.y * inv);
    h[6] = (_Float16)(vb.z * inv); h[7] = (_Float16)(vb.w * inv);
    *(v8h*)(Zh + (size_t)row * DIM + lane * 8) = h;
}

// ---------------------------------------------------------------------------
// Kernel 2: positive-pair logits in full fp32. One wave per pair.
// ---------------------------------------------------------------------------
__global__ __launch_bounds__(256)
void nxtent_simpos_kernel(const float* __restrict__ z1,
                          const float* __restrict__ z2,
                          float* __restrict__ simpos) {
    const int wave = threadIdx.x >> 5;
    const int lane = threadIdx.x & 31;
    const int i    = blockIdx.x * 8 + wave;              // 0..4095

    const float4* a = (const float4*)(z1 + (size_t)i * DIM);
    const float4* b = (const float4*)(z2 + (size_t)i * DIM);

    float s11 = 0.f, s22 = 0.f, s12 = 0.f;
    #pragma unroll
    for (int q = 0; q < 2; ++q) {
        const float4 x = a[lane * 2 + q];
        const float4 y = b[lane * 2 + q];
        s11 += x.x*x.x + x.y*x.y + x.z*x.z + x.w*x.w;
        s22 += y.x*y.x + y.y*y.y + y.z*y.z + y.w*y.w;
        s12 += x.x*y.x + x.y*y.y + x.z*y.z + x.w*y.w;
    }
    #pragma unroll
    for (int off = 1; off < 32; off <<= 1) {
        s11 += __shfl_xor(s11, off, 32);
        s22 += __shfl_xor(s22, off, 32);
        s12 += __shfl_xor(s12, off, 32);
    }
    if (lane == 0) {
        const float d = fmaxf(sqrtf(s11), NORM_EPS) * fmaxf(sqrtf(s22), NORM_EPS);
        simpos[i] = (s12 / d) * INV_T;
    }
}

// ---------------------------------------------------------------------------
// Kernel 3: fused  S = Zh·Zhᵀ/T  ->  exp  ->  per-row sum, LDS-staged.
// Block covers a 128(i) x 128(j) macro-tile. Stage 1: 128 A-rows + 128
// B-rows (full K=256, f16) are copied into LDS ONCE via the CDNA5 async
// copy path (global_load_async_to_lds_b128 / ASYNCcnt), padded 16B/row for
// bank rotation. Stage 2: 8 waves (4x2 grid), each computing a 32x64 region
// = 2x4 WMMA tiles (8 independent accumulators), read fragments from LDS
// (ds_load_b128, <=2-way bank conflicts). Global traffic drops 3x vs the
// register-only version. Diagonal skipped (exp(-1e9)==0). Row partials via
// ds_add_f32, one global atomic per row per block.
// ---------------------------------------------------------------------------
__global__ __launch_bounds__(256)
void nxtent_gemm_exp_kernel(const _Float16* __restrict__ Zh,
                            float* __restrict__ rowsum) {
    __shared__ _Float16 smem[256 * LDS_PITCH_H];          // ~132 KB
    __shared__ float    lsum[128];

    const int wave = threadIdx.x >> 5;
    const int lane = threadIdx.x & 31;
    const int wi   = wave >> 1;                          // 0..3 -> i offset
    const int wj   = wave & 1;                           // 0..1 -> j offset

    const int block_i = blockIdx.x * 128;                // 64 blocks in x
    const int block_j = blockIdx.y * 128;                // 64 blocks in y

    if (threadIdx.x < 128) lsum[threadIdx.x] = 0.0f;

    // ---- Stage A (rows block_i..+127) and B (rows block_j..+127) into LDS.
    // 8192 16-byte chunks, 32 async copies per thread (ASYNCcnt <= 32).
    {
        const uint32_t lds_base = (uint32_t)(uintptr_t)&smem[0];
        #pragma unroll 4
        for (int it = 0; it < 32; ++it) {
            const int c    = threadIdx.x + it * 256;     // 0..8191
            const int row  = c >> 5;                     // 0..255
            const int col  = c & 31;                     // 16B units within row
            const int grow = (row < 128) ? (block_i + row)
                                         : (block_j + (row - 128));
            const _Float16* gp = Zh + (size_t)grow * DIM + col * 8;
            const uint32_t  la = lds_base + (uint32_t)row * (LDS_PITCH_H * 2)
                                          + (uint32_t)col * 16;
            asm volatile("global_load_async_to_lds_b128 %0, %1, off"
                         :: "v"(la), "v"(gp) : "memory");
        }
        asm volatile("s_wait_asynccnt 0x0" ::: "memory");
    }
    __syncthreads();

    const int mn = lane & 15;                            // row (A) / col (B)
    const int g  = lane >> 4;                            // lane half select

    const _Float16* a0L = &smem[(size_t)(wi * 32 + mn) * LDS_PITCH_H];
    const _Float16* a1L = a0L + (size_t)16 * LDS_PITCH_H;
    const _Float16* bL  = &smem[(size_t)(128 + wj * 64 + mn) * LDS_PITCH_H];

    v8f acc[2][4];
    #pragma unroll
    for (int ti = 0; ti < 2; ++ti)
        #pragma unroll
        for (int tj = 0; tj < 4; ++tj)
            acc[ti][tj] = (v8f){};

    #pragma unroll
    for (int kk = 0; kk < DIM; kk += 32) {
        // A frag (16x32 f16): halfs 0-7 -> K = kk + g*8 + t,
        //                     halfs 8-15 -> K = kk + 16 + g*8 + t
        AFrag A0, A1;
        A0.h[0] = *(const v8h*)(a0L + kk + g * 8);
        A0.h[1] = *(const v8h*)(a0L + kk + 16 + g * 8);
        A1.h[0] = *(const v8h*)(a1L + kk + g * 8);
        A1.h[1] = *(const v8h*)(a1L + kk + 16 + g * 8);

        // B frags (32x16 f16): halfs 0-7 -> K = kk + g*8 + t, halfs 8-15 -> K = kk + 16 + g*8 + t (2x b128)
        AFrag B[4];
        #pragma unroll
        for (int tj = 0; tj < 4; ++tj) {
            const _Float16* bt = bL + (size_t)tj * 16 * LDS_PITCH_H + kk + g * 8;
            B[tj].h[0] = *(const v8h*)(bt);
            B[tj].h[1] = *(const v8h*)(bt + 16);
        }

        #pragma unroll
        for (int tj = 0; tj < 4; ++tj) {
            acc[0][tj] = WMMA_F16(A0.v, B[tj].v, acc[0][tj]);
            acc[1][tj] = WMMA_F16(A1.v, B[tj].v, acc[1][tj]);
        }
    }

    // C layout: VGPR r -> M = r + 8*g, N = lane&15
    #pragma unroll
    for (int ti = 0; ti < 2; ++ti) {
        #pragma unroll
        for (int tj = 0; tj < 4; ++tj) {
            const v8f c  = acc[ti][tj];
            const int gj = block_j + wj * 64 + tj * 16 + mn;
            #pragma unroll
            for (int r = 0; r < 8; ++r) {
                const int mrow = r + g * 8;
                const int gi   = block_i + wi * 32 + ti * 16 + mrow;
                float e = (gi == gj) ? 0.0f : __expf(c[r] * INV_T);
                // reduce over the 16 lanes of this half (same output row)
                #pragma unroll
                for (int off = 1; off < 16; off <<= 1)
                    e += __shfl_xor(e, off, 32);
                if (mn == 0)
                    atomicAdd(&lsum[wi * 32 + ti * 16 + mrow], e);  // ds_add_f32
            }
        }
    }
    __syncthreads();

    if (threadIdx.x < 128)
        atomicAdd(&rowsum[block_i + threadIdx.x], lsum[threadIdx.x]);
}

// ---------------------------------------------------------------------------
// Kernel 4: loss = mean_i( log(rowsum[i]) - simpos[i mod N] ), single block.
// ---------------------------------------------------------------------------
__global__ __launch_bounds__(256)
void nxtent_finalize_kernel(const float* __restrict__ rowsum,
                            const float* __restrict__ simpos,
                            float* __restrict__ out) {
    float local = 0.0f;
    for (int i = threadIdx.x; i < TWO_N; i += 256)
        local += __logf(rowsum[i]) - simpos[i & (N_PAIRS - 1)];

    __shared__ float red[256];
    red[threadIdx.x] = local;
    __syncthreads();
    #pragma unroll
    for (int s = 128; s > 0; s >>= 1) {
        if (threadIdx.x < s) red[threadIdx.x] += red[threadIdx.x + s];
        __syncthreads();
    }
    if (threadIdx.x == 0) out[0] = red[0] / (float)TWO_N;
}

// ---------------------------------------------------------------------------
extern "C" void kernel_launch(void* const* d_in, const int* in_sizes, int n_in,
                              void* d_out, int out_size, void* d_ws, size_t ws_size,
                              hipStream_t stream) {
    const float* z1 = (const float*)d_in[0];
    const float* z2 = (const float*)d_in[1];
    // d_in[2], d_in[3] (labels) are unused by the reference forward.
    float* out = (float*)d_out;

    char* ws = (char*)d_ws;
    _Float16* Zh     = (_Float16*)ws;                            // 4 MB
    float*    rowsum = (float*)(ws + (size_t)TWO_N * DIM * 2);   // 32 KB
    float*    simpos = rowsum + TWO_N;                           // 16 KB

    nxtent_zero_kernel<<<(TWO_N + 255) / 256, 256, 0, stream>>>(rowsum, TWO_N);
    nxtent_normalize_kernel<<<TWO_N / 8, 256, 0, stream>>>(z1, z2, Zh);
    nxtent_simpos_kernel<<<N_PAIRS / 8, 256, 0, stream>>>(z1, z2, simpos);

    dim3 grid(TWO_N / 128, TWO_N / 128);                         // 64 x 64
    nxtent_gemm_exp_kernel<<<grid, 256, 0, stream>>>(Zh, rowsum);

    nxtent_finalize_kernel<<<1, 256, 0, stream>>>(rowsum, simpos, out);
}
